// CPNKIQwenAttention_2723009265807
// MI455X (gfx1250) — matrix operand order repaired and checked by reading, not compiled, not run on hardware
//
#include <hip/hip_runtime.h>

typedef __attribute__((ext_vector_type(16))) _Float16 v16h;
typedef __attribute__((ext_vector_type(8)))  _Float16 v8h;
typedef __attribute__((ext_vector_type(8)))  float    v8f;

#define D_MODEL 3072
#define NHEAD   24
#define HD      128
#define S_IMG   2048
#define S_TXT   256
#define S_TOT   2304

__device__ __forceinline__ v16h cat8(v8h lo, v8h hi) {
  return __builtin_shufflevector(lo, hi, 0,1,2,3,4,5,6,7,8,9,10,11,12,13,14,15);
}
__device__ __forceinline__ v8f wmma_f16(v16h a, v16h b, v8f c) {
  // D = A(16x32) * B(32x16) + C, f16 in, f32 accumulate
  return __builtin_amdgcn_wmma_f32_16x16x32_f16(false, a, false, b, (short)0, c,
                                                false, false);
}

// Async global->LDS 16B copy (CDNA5, ASYNCcnt-tracked, no VGPR data).
__device__ __forceinline__ void async_ld_b128(_Float16* ldsdst, const _Float16* gsrc) {
  const unsigned laddr = (unsigned)(unsigned long long)ldsdst;  // low 32b of generic = LDS addr
  asm volatile("global_load_async_to_lds_b128 %0, %1, off"
               :: "v"(laddr), "v"(gsrc)
               : "memory");
}
__device__ __forceinline__ void wait_async0() {
  asm volatile("s_wait_asynccnt 0x0" ::: "memory");
}

// ---------------------------------------------------------------------------
// fp32 -> fp16 conversion (one pass over weights/activations)
// ---------------------------------------------------------------------------
__global__ __launch_bounds__(256) void cvt_f32_f16(const float* __restrict__ src,
                                                   _Float16* __restrict__ dst) {
  const size_t i = ((size_t)blockIdx.x * blockDim.x + threadIdx.x) * 4;
  const float4 v = *(const float4*)(src + i);
  dst[i + 0] = (_Float16)v.x; dst[i + 1] = (_Float16)v.y;
  dst[i + 2] = (_Float16)v.z; dst[i + 3] = (_Float16)v.w;
}

// ---------------------------------------------------------------------------
// GEMM: C[M,N] = X[M,K] @ W[N,K]^T + bias[N]   (f16 in, f32 acc, OutT out)
// Block tile 128x128, K-tile 64, LDS double-buffered via async copies.
// 8 waves: 2(M) x 4(N); each wave 64x32 of C (4x2 frags).
// ---------------------------------------------------------------------------
#define MT  128
#define NT  128
#define KTG 64

template <typename OutT>
__global__ __launch_bounds__(256) void gemm_h_xwt_bias(
    const _Float16* __restrict__ X, const _Float16* __restrict__ W,
    const float* __restrict__ bias, OutT* __restrict__ C,
    int M, int N, int K)
{
  __shared__ __align__(16) _Float16 sX[2][MT][KTG + 8];
  __shared__ __align__(16) _Float16 sW[2][NT][KTG + 8];
  const int tid  = threadIdx.x;
  const int lane = tid & 31, wid = tid >> 5;
  const int wm = wid & 1, wn = wid >> 1;
  const int l16 = lane & 15, half = lane >> 4;
  const int bm = blockIdx.y * MT, bn = blockIdx.x * NT;

  v8f acc[4][2];
  #pragma unroll
  for (int i = 0; i < 4; i++)
    #pragma unroll
    for (int j = 0; j < 2; j++) acc[i][j] = (v8f)0.0f;

  // stage one 128x64-half tile pair: 8 chunks/row, 1024 chunks/tile, 4+4 per thread
  auto issue = [&](int t, int b) {
    const size_t kb = (size_t)t * KTG;
    #pragma unroll
    for (int i = 0; i < 4; i++) {
      const int ch = tid + i * 256;
      const int r = ch >> 3, c = (ch & 7) * 8;
      async_ld_b128(&sX[b][r][c], X + (size_t)(bm + r) * K + kb + c);
      async_ld_b128(&sW[b][r][c], W + (size_t)(bn + r) * K + kb + c);
    }
  };

  const int T = K / KTG;
  issue(0, 0);
  for (int t = 0; t < T; t++) {
    const int cur = t & 1;
    wait_async0();          // my tile-t copies done (t+1 not yet issued)
    __syncthreads();        // all waves' tile-t visible; compute(t-1) finished everywhere
    if (t + 1 < T) issue(t + 1, cur ^ 1);   // DMA overlaps compute below
    #pragma unroll
    for (int kk = 0; kk < KTG; kk += 32) {
      v16h afr[4], bfr[2];
      #pragma unroll
      for (int i = 0; i < 4; i++) {
        const _Float16* base = &sX[cur][wm * 64 + i * 16 + l16][kk + 8 * half];
        afr[i] = cat8(*(const v8h*)base, *(const v8h*)(base + 16));
      }
      #pragma unroll
      for (int j = 0; j < 2; j++) {
        const _Float16* base = &sW[cur][wn * 32 + j * 16 + l16][kk + 8 * half];
        bfr[j] = cat8(*(const v8h*)base, *(const v8h*)(base + 16));
      }
      #pragma unroll
      for (int i = 0; i < 4; i++)
        #pragma unroll
        for (int j = 0; j < 2; j++)
          acc[i][j] = wmma_f16(afr[i], bfr[j], acc[i][j]);
    }
  }
  // epilogue: C layout lane col = l16, rows r + 8*half
  #pragma unroll
  for (int j = 0; j < 2; j++) {
    const int col = bn + wn * 32 + j * 16 + l16;
    const float bv = bias[col];
    #pragma unroll
    for (int i = 0; i < 4; i++)
      #pragma unroll
      for (int r = 0; r < 8; r++) {
        const int row = bm + wm * 64 + i * 16 + r + 8 * half;
        C[(size_t)row * N + col] = (OutT)(acc[i][j][r] + bv);
      }
  }
}

// ---------------------------------------------------------------------------
// Post: RMSNorm + RoPE on q,k; pack q,k -> f16 [h][s][128], v -> f16 [h][128][s]
// Inputs are combined f16 [2304][3072] (txt rows 0..255, img rows 256..2303).
// One wave per (token, head); lane handles 4 head-dim elements.
// ---------------------------------------------------------------------------
__global__ __launch_bounds__(256) void post_qkv(
  const _Float16* __restrict__ qp, const _Float16* __restrict__ kp,
  const _Float16* __restrict__ vp,
  const float* __restrict__ nqw, const float* __restrict__ nkw,
  const float* __restrict__ naqw, const float* __restrict__ nakw,
  const float* __restrict__ icos, const float* __restrict__ isin,
  const float* __restrict__ tcos, const float* __restrict__ tsin,
  _Float16* __restrict__ q16, _Float16* __restrict__ k16, _Float16* __restrict__ v16t)
{
  const int gw   = (blockIdx.x * blockDim.x + threadIdx.x) >> 5;
  const int lane = threadIdx.x & 31;
  const int t = gw / NHEAD, h = gw % NHEAD;
  const bool txt = (t < S_TXT);
  const int trow = txt ? t : (t - S_TXT);
  const _Float16* qsrc = qp + ((size_t)t * D_MODEL + h * HD);
  const _Float16* ksrc = kp + ((size_t)t * D_MODEL + h * HD);
  const _Float16* vsrc = vp + ((size_t)t * D_MODEL + h * HD);
  const float* cb = (txt ? tcos : icos) + (size_t)trow * (HD / 2);
  const float* sb = (txt ? tsin : isin) + (size_t)trow * (HD / 2);
  const float* wqn = txt ? naqw : nqw;
  const float* wkn = txt ? nakw : nkw;

  const int d0 = lane * 4;
  float qv[4], kv[4], vv[4];
  #pragma unroll
  for (int j = 0; j < 4; j++) {
    qv[j] = (float)qsrc[d0 + j]; kv[j] = (float)ksrc[d0 + j]; vv[j] = (float)vsrc[d0 + j];
  }
  float sq = 0.f, sk = 0.f;
  #pragma unroll
  for (int j = 0; j < 4; j++) { sq += qv[j] * qv[j]; sk += kv[j] * kv[j]; }
  #pragma unroll
  for (int m = 16; m >= 1; m >>= 1) { sq += __shfl_xor(sq, m, 32); sk += __shfl_xor(sk, m, 32); }
  const float rq = rsqrtf(sq * (1.0f / HD) + 1e-6f);
  const float rk = rsqrtf(sk * (1.0f / HD) + 1e-6f);
  #pragma unroll
  for (int j = 0; j < 4; j++) { qv[j] *= rq * wqn[d0 + j]; kv[j] *= rk * wkn[d0 + j]; }
  float qo[4], ko[4];
  #pragma unroll
  for (int pr = 0; pr < 2; pr++) {
    const int i = d0 / 2 + pr;
    const float c = cb[i], s = sb[i];
    const float qr = qv[2 * pr], qm = qv[2 * pr + 1];
    const float kr = kv[2 * pr], km = kv[2 * pr + 1];
    qo[2 * pr] = qr * c - qm * s;  qo[2 * pr + 1] = qr * s + qm * c;
    ko[2 * pr] = kr * c - km * s;  ko[2 * pr + 1] = kr * s + km * c;
  }
  _Float16* qdst = q16 + ((size_t)h * S_TOT + t) * HD + d0;
  _Float16* kdst = k16 + ((size_t)h * S_TOT + t) * HD + d0;
  #pragma unroll
  for (int j = 0; j < 4; j++) { qdst[j] = (_Float16)qo[j]; kdst[j] = (_Float16)ko[j]; }
  #pragma unroll
  for (int j = 0; j < 4; j++)
    v16t[((size_t)h * HD + (d0 + j)) * S_TOT + t] = (_Float16)vv[j];
}

// ---------------------------------------------------------------------------
// Flash attention. Block = 8 waves; wave owns 16 q-rows; block covers 128 rows.
// S^T = K*Q^T so S^T's C-frags reinterpret directly as A-frags of P.
// K/V^T tiles double-buffered in LDS via async copies. Output f16 [2304][3072].
// ---------------------------------------------------------------------------
#define KTILE 64

__global__ __launch_bounds__(256) void flash_attn(
  const _Float16* __restrict__ q16, const _Float16* __restrict__ k16,
  const _Float16* __restrict__ v16t, _Float16* __restrict__ out)
{
  __shared__ __align__(16) _Float16 sK[2][KTILE][HD + 8];     // [k][d]
  __shared__ __align__(16) _Float16 sV[2][HD][KTILE + 8];     // [d][k] (V^T)
  const int h = blockIdx.y;
  const int tid = threadIdx.x, lane = tid & 31, wid = tid >> 5;
  const int l16 = lane & 15, half = lane >> 4;
  const int qrow0 = blockIdx.x * 128 + wid * 16;

  // Preload Q as B-matrix frags: lane n = q row (l16), K-dim = d
  v16h qfr[4];
  {
    const _Float16* qb = q16 + ((size_t)h * S_TOT + qrow0 + l16) * HD + 8 * half;
    #pragma unroll
    for (int d = 0; d < 4; d++)
      qfr[d] = cat8(*(const v8h*)(qb + d * 32), *(const v8h*)(qb + d * 32 + 16));
  }

  auto issue = [&](int t, int b) {
    const size_t k0 = (size_t)t * KTILE;
    #pragma unroll
    for (int i = 0; i < 4; i++) {          // K tile: 64 x 128 halves
      const int ch = tid + i * 256;
      const int r = ch >> 4, c = (ch & 15) * 8;
      async_ld_b128(&sK[b][r][c], k16 + ((size_t)h * S_TOT + k0 + r) * HD + c);
    }
    #pragma unroll
    for (int i = 0; i < 4; i++) {          // V^T tile: 128 x 64 halves
      const int ch = tid + i * 256;
      const int r = ch >> 3, c = (ch & 7) * 8;
      async_ld_b128(&sV[b][r][c], v16t + ((size_t)h * HD + r) * S_TOT + k0 + c);
    }
  };

  v8f o[8];
  #pragma unroll
  for (int i = 0; i < 8; i++) o[i] = (v8f)0.0f;
  float m_i = -3.0e38f, l_i = 0.0f;
  const float scale = 0.088388347648318447f;   // 1/sqrt(128)

  const int T = S_TOT / KTILE;
  issue(0, 0);
  for (int t = 0; t < T; t++) {
    const int cur = t & 1;
    wait_async0();
    __syncthreads();
    if (t + 1 < T) issue(t + 1, cur ^ 1);

    // S^T frags (k x q): A = K rows, B = Q
    v8f st[4];
    #pragma unroll
    for (int f = 0; f < 4; f++) {
      st[f] = (v8f)0.0f;
      #pragma unroll
      for (int d = 0; d < 4; d++) {
        const _Float16* kb = &sK[cur][f * 16 + l16][d * 32 + 8 * half];
        v16h a = cat8(*(const v8h*)kb, *(const v8h*)(kb + 16));
        st[f] = wmma_f16(a, qfr[d], st[f]);
      }
    }
    // online softmax: per-lane column q = l16; rows k = 16f + 8*half + r
    float mnew = m_i;
    #pragma unroll
    for (int f = 0; f < 4; f++)
      #pragma unroll
      for (int r = 0; r < 8; r++) mnew = fmaxf(mnew, st[f][r] * scale);
    mnew = fmaxf(mnew, __shfl_xor(mnew, 16, 32));
    const float alpha = __expf(m_i - mnew);
    float lsum = 0.f;
    _Float16 p[4][8];
    #pragma unroll
    for (int f = 0; f < 4; f++)
      #pragma unroll
      for (int r = 0; r < 8; r++) {
        const float e = __expf(st[f][r] * scale - mnew);
        p[f][r] = (_Float16)e;
        lsum += e;
      }
    lsum += __shfl_xor(lsum, 16, 32);
    l_i = l_i * alpha + lsum;
    m_i = mnew;
    // rescale O: row q' = r + 8*half; lane q' holds its alpha
    float ra[8];
    #pragma unroll
    for (int r = 0; r < 8; r++) ra[r] = __shfl(alpha, 8 * half + r, 32);
    #pragma unroll
    for (int db = 0; db < 8; db++)
      #pragma unroll
      for (int r = 0; r < 8; r++) o[db][r] *= ra[r];
    // P*V: A frags come straight from p (C->A register reinterpretation)
    #pragma unroll
    for (int ks = 0; ks < 2; ks++) {
      v16h a;
      #pragma unroll
      for (int j = 0; j < 8; j++) { a[j] = p[2 * ks][j]; a[j + 8] = p[2 * ks + 1][j]; }
      #pragma unroll
      for (int db = 0; db < 8; db++) {
        const _Float16* vb = &sV[cur][db * 16 + l16][ks * 32 + 8 * half];
        v16h b = cat8(*(const v8h*)vb, *(const v8h*)(vb + 16));
        o[db] = wmma_f16(a, b, o[db]);
      }
    }
  }
  float rl[8];
  #pragma unroll
  for (int r = 0; r < 8; r++) rl[r] = __shfl(l_i, 8 * half + r, 32);
  #pragma unroll
  for (int db = 0; db < 8; db++) {
    const int col = h * HD + db * 16 + l16;
    #pragma unroll
    for (int r = 0; r < 8; r++) {
      const int row = qrow0 + 8 * half + r;
      out[(size_t)row * D_MODEL + col] = (_Float16)(o[db][r] / rl[r]);
    }
  }
}

// ---------------------------------------------------------------------------
extern "C" void kernel_launch(void* const* d_in, const int* in_sizes, int n_in,
                              void* d_out, int out_size, void* d_ws, size_t ws_size,
                              hipStream_t stream)
{
  (void)in_sizes; (void)n_in; (void)out_size; (void)ws_size;
  const float* hs   = (const float*)d_in[0];
  const float* ehs  = (const float*)d_in[1];
  const float* wq   = (const float*)d_in[2];
  const float* bq   = (const float*)d_in[3];
  const float* wk   = (const float*)d_in[4];
  const float* bk   = (const float*)d_in[5];
  const float* wv   = (const float*)d_in[6];
  const float* bv   = (const float*)d_in[7];
  const float* awq  = (const float*)d_in[8];
  const float* abq  = (const float*)d_in[9];
  const float* awk  = (const float*)d_in[10];
  const float* abk  = (const float*)d_in[11];
  const float* awv  = (const float*)d_in[12];
  const float* abv  = (const float*)d_in[13];
  const float* wo   = (const float*)d_in[14];
  const float* bo   = (const float*)d_in[15];
  const float* wao  = (const float*)d_in[16];
  const float* bao  = (const float*)d_in[17];
  const float* nqw  = (const float*)d_in[18];
  const float* nkw  = (const float*)d_in[19];
  const float* naqw = (const float*)d_in[20];
  const float* nakw = (const float*)d_in[21];
  const float* icos = (const float*)d_in[22];
  const float* isin = (const float*)d_in[23];
  const float* tcos = (const float*)d_in[24];
  const float* tsin = (const float*)d_in[25];
  float* out = (float*)d_out;

  char* ws = (char*)d_ws;
  const size_t WSZ = (size_t)D_MODEL * D_MODEL * 2;          // 18874368 B (one f16 weight)
  _Float16* wq16  = (_Float16*)(ws + 0 * WSZ);
  _Float16* wk16  = (_Float16*)(ws + 1 * WSZ);
  _Float16* wv16  = (_Float16*)(ws + 2 * WSZ);
  _Float16* awq16 = (_Float16*)(ws + 3 * WSZ);
  _Float16* awk16 = (_Float16*)(ws + 4 * WSZ);
  _Float16* awv16 = (_Float16*)(ws + 5 * WSZ);
  _Float16* wo16  = (_Float16*)(ws + 6 * WSZ);
  _Float16* wao16 = (_Float16*)(ws + 7 * WSZ);
  size_t off = 8 * WSZ;                                      // 150994944
  _Float16* hs16  = (_Float16*)(ws + off); off += (size_t)S_IMG * D_MODEL * 2;   // 12582912
  _Float16* ehs16 = (_Float16*)(ws + off); off += (size_t)S_TXT * D_MODEL * 2;   // 1572864
  const size_t PSZ = (size_t)S_TOT * D_MODEL * 2;            // 14155776 (combined f16 act)
  _Float16* qp16  = (_Float16*)(ws + off);                   // also reused as attn16 later
  _Float16* kp16  = (_Float16*)(ws + off + PSZ);
  _Float16* vp16  = (_Float16*)(ws + off + 2 * PSZ);
  _Float16* q16   = (_Float16*)(ws + off + 3 * PSZ);
  _Float16* k16   = (_Float16*)(ws + off + 4 * PSZ);
  _Float16* v16t  = (_Float16*)(ws + off + 5 * PSZ);
  _Float16* attn16 = qp16;   // qp16 dead after post_qkv

  const dim3 blk(256);
  const int WBLK = (D_MODEL * D_MODEL) / 1024;   // 9216
  cvt_f32_f16<<<WBLK, blk, 0, stream>>>(wq,  wq16);
  cvt_f32_f16<<<WBLK, blk, 0, stream>>>(wk,  wk16);
  cvt_f32_f16<<<WBLK, blk, 0, stream>>>(wv,  wv16);
  cvt_f32_f16<<<WBLK, blk, 0, stream>>>(awq, awq16);
  cvt_f32_f16<<<WBLK, blk, 0, stream>>>(awk, awk16);
  cvt_f32_f16<<<WBLK, blk, 0, stream>>>(awv, awv16);
  cvt_f32_f16<<<WBLK, blk, 0, stream>>>(wo,  wo16);
  cvt_f32_f16<<<WBLK, blk, 0, stream>>>(wao, wao16);
  cvt_f32_f16<<<(S_IMG * D_MODEL) / 1024, blk, 0, stream>>>(hs,  hs16);
  cvt_f32_f16<<<(S_TXT * D_MODEL) / 1024, blk, 0, stream>>>(ehs, ehs16);

  // QKV projections -> combined f16 activations (txt rows 0..255, img rows 256..)
  _Float16* qp_img = qp16 + (size_t)S_TXT * D_MODEL;
  _Float16* kp_img = kp16 + (size_t)S_TXT * D_MODEL;
  _Float16* vp_img = vp16 + (size_t)S_TXT * D_MODEL;
  gemm_h_xwt_bias<_Float16><<<dim3(24, 16), blk, 0, stream>>>(hs16,  wq16,  bq,  qp_img, S_IMG, D_MODEL, D_MODEL);
  gemm_h_xwt_bias<_Float16><<<dim3(24, 16), blk, 0, stream>>>(hs16,  wk16,  bk,  kp_img, S_IMG, D_MODEL, D_MODEL);
  gemm_h_xwt_bias<_Float16><<<dim3(24, 16), blk, 0, stream>>>(hs16,  wv16,  bv,  vp_img, S_IMG, D_MODEL, D_MODEL);
  gemm_h_xwt_bias<_Float16><<<dim3(24, 2),  blk, 0, stream>>>(ehs16, awq16, abq, qp16,   S_TXT, D_MODEL, D_MODEL);
  gemm_h_xwt_bias<_Float16><<<dim3(24, 2),  blk, 0, stream>>>(ehs16, awk16, abk, kp16,   S_TXT, D_MODEL, D_MODEL);
  gemm_h_xwt_bias<_Float16><<<dim3(24, 2),  blk, 0, stream>>>(ehs16, awv16, abv, vp16,   S_TXT, D_MODEL, D_MODEL);

  post_qkv<<<dim3(6912), blk, 0, stream>>>(qp16, kp16, vp16, nqw, nkw, naqw, nakw,
                                           icos, isin, tcos, tsin, q16, k16, v16t);
  flash_attn<<<dim3(18, 24), blk, 0, stream>>>(q16, k16, v16t, attn16);

  gemm_h_xwt_bias<float><<<dim3(24, 16), blk, 0, stream>>>(attn16 + (size_t)S_TXT * D_MODEL, wo16, bo,
                                                           out, S_IMG, D_MODEL, D_MODEL);
  gemm_h_xwt_bias<float><<<dim3(24, 2),  blk, 0, stream>>>(attn16, wao16, bao,
                                                           out + (size_t)S_IMG * D_MODEL, S_TXT, D_MODEL, D_MODEL);
}
